// ContextBottleneck_85822036509275
// MI455X (gfx1250) — matrix operand order repaired and checked by reading, not compiled, not run on hardware
//
#include <hip/hip_runtime.h>
#include <hip/hip_bf16.h>

// ---------------------------------------------------------------------------
// ContextBottleneck fused kernel for gfx1250 (MI455X)
//   out = hidden + alpha * (Up(SiLU(Down(LN(hidden)))) - hidden)
// B=4, S=4096, D=2048, Db=512  ->  16384 rows.
// GEMMs run on v_wmma_f32_16x16x32_bf16; everything else fused around them.
// ---------------------------------------------------------------------------

typedef __attribute__((ext_vector_type(16))) __bf16 v16bf;
typedef __attribute__((ext_vector_type(8)))  float  v8f;

#define D_DIM   2048
#define DB_DIM  512
#define TILE_M  16
#define LN_EPS  1e-5f

static __device__ __forceinline__ unsigned short f2bf(float f) {
  // round-to-nearest-even fp32 -> bf16
  unsigned int u = __builtin_bit_cast(unsigned int, f);
  u += 0x7FFFu + ((u >> 16) & 1u);
  return (unsigned short)(u >> 16);
}

struct B32x8 { uint4 lo, hi; };  // 32 bytes == one v16bf fragment

// A-fragment (16x32 bf16, row-major in LDS with stride ldA elements).
// Lane layout (ISA 7.12.2): M = lane&15; lanes 0-15 hold K = {0..7,16..23},
// lanes 16-31 hold K = {8..15,24..31}. Each half is 8 contiguous K values ->
// two 16-byte LDS loads per lane.
static __device__ __forceinline__ v16bf load_a_frag(const unsigned short* rowbase,
                                                    int k0, int hi) {
  B32x8 t;
  t.lo = *(const uint4*)(rowbase + k0 + hi * 8);
  t.hi = *(const uint4*)(rowbase + k0 + 16 + hi * 8);
  return __builtin_bit_cast(v16bf, t);
}

// B-fragment (32x16 bf16) from a column-major (transposed) bf16 weight
// matrix wt[N][Kd]. Lane layout: N = lane&15; lanes 0-15 hold K = {0..7,16..23},
// lanes 16-31 hold K = {8..15,24..31} -> two 16-byte loads per lane.
static __device__ __forceinline__ v16bf load_b_frag(const unsigned short* wt,
                                                    int n, int Kd, int k0, int hi) {
  const unsigned short* p = wt + (size_t)n * Kd + k0 + hi * 8;
  B32x8 t;
  t.lo = *(const uint4*)(p);
  t.hi = *(const uint4*)(p + 16);
  return __builtin_bit_cast(v16bf, t);
}

// ---------------------------------------------------------------------------
// Prep: dst[n*K + k] = bf16(src[k*N + n])  (convert + transpose)
// Each thread handles 8 consecutive k for one n: coalesced reads,
// 16-byte contiguous writes.
// ---------------------------------------------------------------------------
__global__ void convT_kernel(const float* __restrict__ src,
                             unsigned short* __restrict__ dst,
                             int K, int N) {
  int t = blockIdx.x * blockDim.x + threadIdx.x;
  int total = (K / 8) * N;
  if (t >= total) return;
  int k0 = (t / N) * 8;
  int n  = t % N;
  ushort4 a, b;
  a.x = f2bf(src[(size_t)(k0 + 0) * N + n]);
  a.y = f2bf(src[(size_t)(k0 + 1) * N + n]);
  a.z = f2bf(src[(size_t)(k0 + 2) * N + n]);
  a.w = f2bf(src[(size_t)(k0 + 3) * N + n]);
  b.x = f2bf(src[(size_t)(k0 + 4) * N + n]);
  b.y = f2bf(src[(size_t)(k0 + 5) * N + n]);
  b.z = f2bf(src[(size_t)(k0 + 6) * N + n]);
  b.w = f2bf(src[(size_t)(k0 + 7) * N + n]);
  unsigned short* d = dst + (size_t)n * K + k0;
  *(ushort4*)(d)     = a;
  *(ushort4*)(d + 4) = b;
}

// ---------------------------------------------------------------------------
// Main fused kernel: one 16-row tile per workgroup, 256 threads = 8 waves.
// ---------------------------------------------------------------------------
__global__ void __launch_bounds__(256)
bottleneck_kernel(const float* __restrict__ hidden,
                  const float* __restrict__ gamma,
                  const float* __restrict__ beta,
                  const unsigned short* __restrict__ WdT,  // [DB_DIM][D_DIM] bf16
                  const float* __restrict__ b_down,
                  const unsigned short* __restrict__ WuT,  // [D_DIM][DB_DIM] bf16
                  const float* __restrict__ b_up,
                  const float* __restrict__ alpha_p,
                  float* __restrict__ out) {
  extern __shared__ char smem[];
  unsigned short* sh_hn = (unsigned short*)smem;                         // [16][2048] bf16
  unsigned short* sh_z  = (unsigned short*)(smem + TILE_M * D_DIM * 2);  // [16][512]  bf16

  const int tid  = threadIdx.x;
  const int wave = tid >> 5;
  const int lane = tid & 31;
  const int l15  = lane & 15;
  const int hi   = lane >> 4;     // half-wave id
  const int row0 = blockIdx.x * TILE_M;

  const float alpha = alpha_p[0];

  // ---------------- Phase 1: LayerNorm (2 rows per wave) -> bf16 LDS -------
  #pragma unroll
  for (int rr = 0; rr < 2; ++rr) {
    const int r = wave * 2 + rr;
    const float* hrow = hidden + (size_t)(row0 + r) * D_DIM;
    float s = 0.f, sq = 0.f;
    #pragma unroll
    for (int i = 0; i < D_DIM / 128; ++i) {
      float4 v = ((const float4*)hrow)[lane + i * 32];
      s  += v.x + v.y + v.z + v.w;
      sq += v.x * v.x + v.y * v.y + v.z * v.z + v.w * v.w;
    }
    #pragma unroll
    for (int off = 16; off > 0; off >>= 1) {
      s  += __shfl_xor(s,  off, 32);
      sq += __shfl_xor(sq, off, 32);
    }
    const float mu   = s * (1.0f / D_DIM);
    const float var  = sq * (1.0f / D_DIM) - mu * mu;
    const float rstd = rsqrtf(var + LN_EPS);
    unsigned short* drow = sh_hn + r * D_DIM;
    #pragma unroll
    for (int i = 0; i < D_DIM / 128; ++i) {
      const int i4 = lane + i * 32;
      float4 v  = ((const float4*)hrow)[i4];
      float4 g  = ((const float4*)gamma)[i4];
      float4 bt = ((const float4*)beta)[i4];
      ushort4 o;
      o.x = f2bf((v.x - mu) * rstd * g.x + bt.x);
      o.y = f2bf((v.y - mu) * rstd * g.y + bt.y);
      o.z = f2bf((v.z - mu) * rstd * g.z + bt.z);
      o.w = f2bf((v.w - mu) * rstd * g.w + bt.w);
      *(ushort4*)(drow + i4 * 4) = o;
    }
  }
  __syncthreads();

  // ---------------- Phase 2: GEMM1 (16x2048 @ 2048x512) + SiLU -> z --------
  {
    const unsigned short* arow = sh_hn + (size_t)l15 * D_DIM;  // A row: M = lane&15
    const int nbase = wave * 64;                               // 4 N-tiles per wave
    v8f acc0 = {0.f,0.f,0.f,0.f,0.f,0.f,0.f,0.f};
    v8f acc[4] = {acc0, acc0, acc0, acc0};
    for (int k0 = 0; k0 < D_DIM; k0 += 32) {
      v16bf a = load_a_frag(arow, k0, hi);
      #pragma unroll
      for (int t = 0; t < 4; ++t) {
        v16bf b = load_b_frag(WdT, nbase + t * 16 + l15, D_DIM, k0, hi);
        acc[t] = __builtin_amdgcn_wmma_f32_16x16x32_bf16(
            false, a, false, b, (short)0, acc[t], false, false);
      }
    }
    // epilogue: + b_down, SiLU, store bf16 z tile
    #pragma unroll
    for (int t = 0; t < 4; ++t) {
      const int n = nbase + t * 16 + l15;
      const float bd = b_down[n];
      #pragma unroll
      for (int r = 0; r < 8; ++r) {
        const int m = r + hi * 8;                 // C layout: lanes16-31 -> M=r+8
        const float v  = acc[t][r] + bd;
        const float sv = v / (1.0f + __expf(-v)); // SiLU
        sh_z[m * DB_DIM + n] = f2bf(sv);
      }
    }
  }
  __syncthreads();

  // ---------------- Phase 3: GEMM2 (16x512 @ 512x2048) + residual ----------
  {
    const unsigned short* zrow = sh_z + (size_t)l15 * DB_DIM;
    #pragma unroll
    for (int pass = 0; pass < 4; ++pass) {
      const int nbase = wave * 256 + pass * 64;
      v8f acc0 = {0.f,0.f,0.f,0.f,0.f,0.f,0.f,0.f};
      v8f acc[4] = {acc0, acc0, acc0, acc0};
      #pragma unroll
      for (int k0 = 0; k0 < DB_DIM; k0 += 32) {
        v16bf a = load_a_frag(zrow, k0, hi);
        #pragma unroll
        for (int t = 0; t < 4; ++t) {
          v16bf b = load_b_frag(WuT, nbase + t * 16 + l15, DB_DIM, k0, hi);
          acc[t] = __builtin_amdgcn_wmma_f32_16x16x32_bf16(
              false, a, false, b, (short)0, acc[t], false, false);
        }
      }
      #pragma unroll
      for (int t = 0; t < 4; ++t) {
        const int n = nbase + t * 16 + l15;
        const float bu = b_up[n];
        #pragma unroll
        for (int r = 0; r < 8; ++r) {
          const int m = r + hi * 8;
          const size_t idx = (size_t)(row0 + m) * D_DIM + n;
          const float h = hidden[idx];
          const float v = acc[t][r] + bu;
          out[idx] = h + alpha * (v - h);
        }
      }
    }
  }
}

// ---------------------------------------------------------------------------
extern "C" void kernel_launch(void* const* d_in, const int* in_sizes, int n_in,
                              void* d_out, int out_size, void* d_ws, size_t ws_size,
                              hipStream_t stream) {
  const float* hidden = (const float*)d_in[0];
  const float* gamma  = (const float*)d_in[1];
  const float* beta   = (const float*)d_in[2];
  const float* W_down = (const float*)d_in[3];   // [D][Db] row-major
  const float* b_down = (const float*)d_in[4];
  const float* W_up   = (const float*)d_in[5];   // [Db][D] row-major
  const float* b_up   = (const float*)d_in[6];
  const float* alpha  = (const float*)d_in[7];
  float* out = (float*)d_out;

  const int rows = in_sizes[0] / D_DIM;          // B*S = 16384

  // workspace: bf16 transposed weights
  unsigned short* WdT = (unsigned short*)d_ws;                   // [Db][D]  2 MB
  unsigned short* WuT = WdT + (size_t)D_DIM * DB_DIM;            // [D][Db]  2 MB

  // prep: convert + transpose both weights
  {
    const int tot = (D_DIM / 8) * DB_DIM;   // == (DB_DIM/8)*D_DIM == 131072
    convT_kernel<<<(tot + 255) / 256, 256, 0, stream>>>(W_down, WdT, D_DIM, DB_DIM);
    convT_kernel<<<(tot + 255) / 256, 256, 0, stream>>>(W_up,   WuT, DB_DIM, D_DIM);
  }

  const size_t shmem = (size_t)TILE_M * D_DIM * 2 + (size_t)TILE_M * DB_DIM * 2; // 80 KB
  hipFuncSetAttribute((const void*)bottleneck_kernel,
                      hipFuncAttributeMaxDynamicSharedMemorySize, (int)shmem);

  bottleneck_kernel<<<rows / TILE_M, 256, shmem, stream>>>(
      hidden, gamma, beta, WdT, b_down, WuT, b_up, alpha, out);
}
